// TopKGating_16887811408078
// MI455X (gfx1250) — compile-verified, matrix operand order checked by reading, not run on hardware
//
#include <hip/hip_runtime.h>
#include <hip/hip_bf16.h>

// ---------------------------------------------------------------------------
// MoE Top-K gating for MI455X (gfx1250, wave32, WMMA).
//   logits[t,e] = sum_d x[t,d]*W[e,d] + b[e]   (t=0..16383, e=0..63)
//   -> top-2 softmax gate scores + indices, plus KL load-balance aux loss.
//
// GEMM: bf16 V_WMMA_F32_16X16X32_BF16 (fp32 accumulate), x converted on the
// fly, W pre-converted+pre-swizzled to bf16 B-fragment blobs in workspace.
// One token tile (16 tokens x 64 experts) per block; D=4096 K-split across
// 4 waves (-> 4096 waves for latency hiding), combined via LDS.
// ---------------------------------------------------------------------------

#define BB 4
#define SS 4096
#define DD 4096
#define EE 64
#define NTOK (BB * SS)            // 16384 tokens
#define DSPLIT 4                  // K-split waves per token tile
#define KSTEPS (DD / 32 / DSPLIT) // 32 k-steps of 32 per wave
#define BLOCK (DSPLIT * 32)       // 128 threads
#define NBLOCKS (NTOK / 16)       // 1024 blocks, one 16-token tile each
#define LDSW 68                   // padded row stride (floats), conflict-free
#define WBF_BYTES (EE * DD * 2)   // 512 KB pre-converted W in workspace

typedef __attribute__((ext_vector_type(16))) __bf16 v16bf;
typedef __attribute__((ext_vector_type(8)))  float  v8f;
typedef __attribute__((ext_vector_type(4)))  float  v4f;

// A-fragment for V_WMMA_F32_16X16X32_BF16 from an fp32 row.
// ISA 16-bit 16x32 layout: lanes 0-15 hold K={0..7,16..23}, lanes 16-31
// K={8..15,24..31}; VGPR i packs K-pair (2i,2i+1) of its group.
__device__ __forceinline__ v16bf make_frag(const float* __restrict__ row, int kh) {
  const float* g0 = row + kh * 8;
  const float* g1 = row + 16 + kh * 8;
  v4f a = *(const v4f*)(g0);
  v4f b = *(const v4f*)(g0 + 4);
  v4f c = *(const v4f*)(g1);
  v4f d = *(const v4f*)(g1 + 4);
  v16bf f;
  f[0]  = (__bf16)a[0]; f[1]  = (__bf16)a[1]; f[2]  = (__bf16)a[2]; f[3]  = (__bf16)a[3];
  f[4]  = (__bf16)b[0]; f[5]  = (__bf16)b[1]; f[6]  = (__bf16)b[2]; f[7]  = (__bf16)b[3];
  f[8]  = (__bf16)c[0]; f[9]  = (__bf16)c[1]; f[10] = (__bf16)c[2]; f[11] = (__bf16)c[3];
  f[12] = (__bf16)d[0]; f[13] = (__bf16)d[1]; f[14] = (__bf16)d[2]; f[15] = (__bf16)d[3];
  return f;
}

// Pre-convert W[64][4096] fp32 -> bf16 blobs in B-fragment order.
// Blob index = ((kstep*2 + kh)*64 + e); each blob = 16 contiguous bf16 (32B)
// holding K = {kstep*32+kh*8+0..7, kstep*32+16+kh*8+0..7} of expert e.
__global__ __launch_bounds__(256)
void wconv_kernel(const float* __restrict__ W, __bf16* __restrict__ Wbf) {
  int tid   = blockIdx.x * 256 + threadIdx.x;   // 0..16383 == blob index
  int e     = tid & 63;
  int khk   = tid >> 6;        // kstep*2 + kh
  int kh    = khk & 1;
  int kstep = khk >> 1;
  const float* src = W + (size_t)e * DD + kstep * 32 + kh * 8;
  v16bf f;
#pragma unroll
  for (int j = 0; j < 8; ++j) f[j] = (__bf16)src[j];
#pragma unroll
  for (int j = 0; j < 8; ++j) f[8 + j] = (__bf16)src[16 + j];
  *(v16bf*)(Wbf + (size_t)tid * 16) = f;
}

__global__ __launch_bounds__(BLOCK)
void moe_gate_kernel(const float* __restrict__ x,     // [NTOK, DD] fp32
                     const __bf16* __restrict__ Wbf,  // swizzled bf16 W
                     const float* __restrict__ bias,  // [EE]
                     float* __restrict__ out,         // gates | idx | aux
                     float* __restrict__ partial)     // [NBLOCKS] aux partials
{
  __shared__ float lds[DSPLIT][16 * LDSW];

  const int lane = threadIdx.x & 31;
  const int wave = threadIdx.x >> 5;        // K-slice id 0..3
  const int t0   = blockIdx.x * 16;         // first token of this tile
  const int m    = lane & 15;               // token row / expert col in tile
  const int kh   = lane >> 4;               // K-half select

  const float* __restrict__ xrow = x + (size_t)(t0 + m) * DD + wave * (DD / DSPLIT);
  const int ks0 = wave * KSTEPS;            // first global k-step of slice

  v8f acc0 = {}, acc1 = {}, acc2 = {}, acc3 = {};

  for (int s = 0; s < KSTEPS; ++s) {
    __builtin_prefetch(xrow + s * 32 + 256, 0, 1);

    v16bf af = make_frag(xrow + s * 32, kh);   // shared across 4 N-tiles

    const __bf16* wp = Wbf + (((size_t)(ks0 + s) * 2 + kh) * 64 + m) * 16;
    v16bf bf0 = *(const v16bf*)(wp);          // experts  0..15
    v16bf bf1 = *(const v16bf*)(wp + 256);    // experts 16..31
    v16bf bf2 = *(const v16bf*)(wp + 512);    // experts 32..47
    v16bf bf3 = *(const v16bf*)(wp + 768);    // experts 48..63

    acc0 = __builtin_amdgcn_wmma_f32_16x16x32_bf16(false, af, false, bf0,
                                                   (short)0, acc0, false, false);
    acc1 = __builtin_amdgcn_wmma_f32_16x16x32_bf16(false, af, false, bf1,
                                                   (short)0, acc1, false, false);
    acc2 = __builtin_amdgcn_wmma_f32_16x16x32_bf16(false, af, false, bf2,
                                                   (short)0, acc2, false, false);
    acc3 = __builtin_amdgcn_wmma_f32_16x16x32_bf16(false, af, false, bf3,
                                                   (short)0, acc3, false, false);
  }

  // ---- spill K-slice partial logits to LDS (C layout: lane=N, vgpr r=row) ----
  const int rbase = kh * 8;   // lanes 0-15: rows 0-7; lanes 16-31: rows 8-15
#pragma unroll
  for (int r = 0; r < 8; ++r) {
    float* dst = &lds[wave][(rbase + r) * LDSW];
    dst[m +  0] = acc0[r];
    dst[m + 16] = acc1[r];
    dst[m + 32] = acc2[r];
    dst[m + 48] = acc3[r];
  }
  __syncthreads();

  // ---- per-token: combine 4 K-slices, +bias, top-2 softmax, lse, aux ----
  float c_t = 0.0f;
  if (threadIdx.x < 16) {
    const int t = threadIdx.x;
    const float* r0 = &lds[0][t * LDSW];
    const float* r1 = &lds[1][t * LDSW];
    const float* r2 = &lds[2][t * LDSW];
    const float* r3 = &lds[3][t * LDSW];

    float v1 = -3.402823466e+38f, v2 = -3.402823466e+38f;
    int   i1 = 0, i2 = 0;
    float slog = 0.0f;
    float logit[EE];
#pragma unroll
    for (int e4 = 0; e4 < EE / 4; ++e4) {
      v4f a = *(const v4f*)(r0 + e4 * 4);
      v4f b = *(const v4f*)(r1 + e4 * 4);
      v4f c = *(const v4f*)(r2 + e4 * 4);
      v4f d = *(const v4f*)(r3 + e4 * 4);
      v4f bb = *(const v4f*)(bias + e4 * 4);
#pragma unroll
      for (int j = 0; j < 4; ++j) {
        int   e = e4 * 4 + j;
        float v = a[j] + b[j] + c[j] + d[j] + bb[j];
        logit[e] = v;
        slog += v;
        if (v > v1)      { v2 = v1; i2 = i1; v1 = v; i1 = e; }  // stable ties:
        else if (v > v2) { v2 = v;  i2 = e; }                   // keep earliest
      }
    }
    float sexp = 0.0f;
#pragma unroll
    for (int e = 0; e < EE; ++e) sexp += __expf(logit[e] - v1);
    float lse = v1 + __logf(sexp);

    float g0 = 1.0f / (1.0f + __expf(v2 - v1));   // softmax over [v1, v2]

    int tg = t0 + t;
    out[2 * tg + 0] = g0;
    out[2 * tg + 1] = 1.0f - g0;
    int* idxp = (int*)(out + 2 * NTOK);           // int32 indices, bit-stored
    idxp[2 * tg + 0] = i1;
    idxp[2 * tg + 1] = i2;

    // sum_e (1/E)(log(1/E) - logp_e) = log(1/E) + lse - mean_e(logit)
    c_t = -4.1588830833596715f /*log(1/64)*/ + lse - slog * (1.0f / EE);
  }

  // wave-0 sum of 16 token terms (other waves compute but never write)
#pragma unroll
  for (int off = 16; off >= 1; off >>= 1) c_t += __shfl_xor(c_t, off, 32);
  if (threadIdx.x == 0) partial[blockIdx.x] = c_t;
}

// Deterministic fixed-order reduction of aux partials (no float atomics ->
// bitwise reproducible across graph replays).
__global__ __launch_bounds__(256)
void aux_reduce_kernel(const float* __restrict__ partial, float* __restrict__ aux_out) {
  __shared__ float s[256];
  float acc = 0.0f;
  for (int i = threadIdx.x; i < NBLOCKS; i += 256) acc += partial[i];
  s[threadIdx.x] = acc;
  __syncthreads();
#pragma unroll
  for (int o = 128; o >= 1; o >>= 1) {
    if (threadIdx.x < o) s[threadIdx.x] += s[threadIdx.x + o];
    __syncthreads();
  }
  if (threadIdx.x == 0)
    aux_out[0] = 0.01f * s[0] * (1.0f / ((float)NTOK * (float)EE));
}

extern "C" void kernel_launch(void* const* d_in, const int* in_sizes, int n_in,
                              void* d_out, int out_size, void* d_ws, size_t ws_size,
                              hipStream_t stream) {
  const float* x    = (const float*)d_in[0];   // [4,4096,4096]
  const float* W    = (const float*)d_in[1];   // [64,4096]
  const float* bias = (const float*)d_in[2];   // [64]
  float* out = (float*)d_out;                  // 32768 gates | 32768 idx | 1 aux

  __bf16* Wbf    = (__bf16*)d_ws;                          // 512 KB
  float*  partial = (float*)((char*)d_ws + WBF_BYTES);     // NBLOCKS floats

  wconv_kernel<<<dim3(EE * DD / 16 / 256), dim3(256), 0, stream>>>(W, Wbf);
  moe_gate_kernel<<<dim3(NBLOCKS), dim3(BLOCK), 0, stream>>>(x, Wbf, bias, out, partial);
  aux_reduce_kernel<<<dim3(1), dim3(256), 0, stream>>>(partial, out + 4 * NTOK);
}
